// GIB_GCN_35021163332284
// MI455X (gfx1250) — matrix operand order repaired and checked by reading, not compiled, not run on hardware
//
#include <hip/hip_runtime.h>
#include <hip/hip_bf16.h>
#include <math.h>

// GIB-GCN pipeline for MI455X (gfx1250, wave32).
// Dense GEMMs -> v_wmma_f32_16x16x32_f16 (f16 in, f32 acc), 16x(16*CT) tile/wave.
// SpMM -> edge-parallel float4 global_atomic_add_f32.
#define FIN  128
#define FHID 256
#define FOUT 40

typedef __attribute__((ext_vector_type(16))) _Float16 v16h;
typedef __attribute__((ext_vector_type(8)))  _Float16 v8h;
typedef __attribute__((ext_vector_type(8)))  float    v8f;

// -------------------------------------------------------------------------
__global__ void gib_zero_kernel(float* __restrict__ p, long long n) {
  long long i = (long long)blockIdx.x * blockDim.x + threadIdx.x;
  if (i < n) p[i] = 0.0f;
}

// -------------------------------------------------------------------------
// One-time: Wt[c * k_in + kk] = (f16) W[kk * n_cols + c]  (transpose+convert)
// Makes B fragments contiguous 16-half runs (two b128 loads per lane).
__global__ void gib_w2h_kernel(const float* __restrict__ W,
                               _Float16* __restrict__ Wt,
                               int k_in, int n_cols) {
  int i = blockIdx.x * blockDim.x + threadIdx.x;
  int total = k_in * n_cols;
  if (i >= total) return;
  int c  = i / k_in;
  int kk = i - c * k_in;
  Wt[i] = (_Float16)W[(size_t)kk * n_cols + c];
}

// -------------------------------------------------------------------------
// GEMM: C[n_rows x n_cols] = A[n_rows x k_in] * W[k_in x n_cols] (+bias)(+relu)
// 8 waves / block; each wave owns a 16 x (16*CT) output tile, sharing one A
// fragment across CT WMMAs per K-step. All loads branchless (clamped
// indices); stores guarded. EXEC stays all-1s around every WMMA.
template <int CT>
__global__ __launch_bounds__(256) void gib_gemm_wmma_kernel(
    const float* __restrict__ A, const _Float16* __restrict__ Wt,
    const float* __restrict__ bias, float* __restrict__ C,
    int n_rows, int k_in, int n_cols, int do_relu) {
  const int lane    = threadIdx.x & 31;
  const int wave    = threadIdx.x >> 5;
  const int rowTile = blockIdx.x * 8 + wave;
  const int r0      = rowTile * 16;
  if (r0 >= n_rows) return;                 // wave-uniform exit

  const int lane15 = lane & 15;
  const int hi     = lane >> 4;

  // clamped row for loads (N % 16 == 0 in practice; clamp keeps it safe)
  int rA = r0 + lane15;
  rA = rA < n_rows ? rA : (n_rows - 1);
  const float* __restrict__ arow = A + (size_t)rA * k_in;

  // per-tile columns; clamp for loads, guard stores
  int  ccol[CT];
  const _Float16* __restrict__ wrow[CT];
#pragma unroll
  for (int t = 0; t < CT; ++t) {
    ccol[t] = blockIdx.y * (16 * CT) + t * 16 + lane15;
    int cl  = ccol[t] < n_cols ? ccol[t] : (n_cols - 1);
    wrow[t] = Wt + (size_t)cl * k_in;
  }

  v8f acc[CT];
#pragma unroll
  for (int t = 0; t < CT; ++t) acc[t] = (v8f){};

  for (int k = 0; k < k_in; k += 32) {
    // A fragment 16x32 f16: lane=row; a[i] -> K=k+hi*16+i.
    // One contiguous 16-float run -> 4x b128.
    const float4 a0 = *(const float4*)(arow + k + hi * 16);
    const float4 a1 = *(const float4*)(arow + k + hi * 16 + 4);
    const float4 a2 = *(const float4*)(arow + k + hi * 16 + 8);
    const float4 a3 = *(const float4*)(arow + k + hi * 16 + 12);
    v16h a;
    a[0]  = (_Float16)a0.x; a[1]  = (_Float16)a0.y;
    a[2]  = (_Float16)a0.z; a[3]  = (_Float16)a0.w;
    a[4]  = (_Float16)a1.x; a[5]  = (_Float16)a1.y;
    a[6]  = (_Float16)a1.z; a[7]  = (_Float16)a1.w;
    a[8]  = (_Float16)a2.x; a[9]  = (_Float16)a2.y;
    a[10] = (_Float16)a2.z; a[11] = (_Float16)a2.w;
    a[12] = (_Float16)a3.x; a[13] = (_Float16)a3.y;
    a[14] = (_Float16)a3.z; a[15] = (_Float16)a3.w;

#pragma unroll
    for (int t = 0; t < CT; ++t) {
      // B fragment 32x16 f16: lane=col; element i -> K=k+hi*16+i,
      // contiguous in transposed Wt -> two b128 loads.
      const _Float16* wp = wrow[t] + k + hi * 16;
      v8h b0 = *(const v8h*)(wp);
      v8h b1 = *(const v8h*)(wp + 8);
      v16h b = __builtin_shufflevector(b0, b1, 0, 1, 2, 3, 4, 5, 6, 7,
                                       8, 9, 10, 11, 12, 13, 14, 15);
      acc[t] = __builtin_amdgcn_wmma_f32_16x16x32_f16(
          false, a, false, b, (short)0, acc[t], false, false);
    }
  }

  // Epilogue: VGPR j -> row r0 + hi*8 + j, col = ccol[t].
#pragma unroll
  for (int t = 0; t < CT; ++t) {
    if (ccol[t] < n_cols) {
      float bv = (bias != nullptr) ? bias[ccol[t]] : 0.0f;
#pragma unroll
      for (int j = 0; j < 8; ++j) {
        int row = r0 + hi * 8 + j;
        if (row < n_rows) {
          float v = acc[t][j] + bv;
          if (do_relu) v = v > 0.0f ? v : 0.0f;
          C[(size_t)row * n_cols + ccol[t]] = v;
        }
      }
    }
  }
}

// -------------------------------------------------------------------------
// SpMM accumulate: Y[row[e],:] += val[e] * X[col[e],:], F = 4*f4 features.
__global__ void gib_spmm_atomic_kernel(
    const int* __restrict__ erow, const int* __restrict__ ecol,
    const float* __restrict__ eval_, const float* __restrict__ X,
    float* __restrict__ Y, long long n_edges, int f4) {
  long long idx   = (long long)blockIdx.x * blockDim.x + threadIdx.x;
  long long total = n_edges * (long long)f4;
  if (idx >= total) return;
  long long e = idx / f4;
  int       g = (int)(idx - e * f4);
  int   r = erow[e];
  int   c = ecol[e];
  float v = eval_[e];
  const int F = f4 * 4;
  const float4 xv = ((const float4*)(X + (size_t)c * F))[g];
  float* yp = Y + (size_t)r * F + (size_t)g * 4;
  atomicAdd(yp + 0, v * xv.x);
  atomicAdd(yp + 1, v * xv.y);
  atomicAdd(yp + 2, v * xv.z);
  atomicAdd(yp + 3, v * xv.w);
}

// -------------------------------------------------------------------------
// VAE sample + KL partial. One block (128 threads, 4 waves) per node.
// term = 0.5*(z^2 - eps^2) - log(std)   (LOG_SQRT_2PI cancels in logq-logp)
__global__ __launch_bounds__(128) void gib_vae_kernel(
    const float* __restrict__ zxhat, const float* __restrict__ eps,
    float* __restrict__ z_out, float* __restrict__ l2_acc, int n_nodes) {
  __shared__ float partial[4];
  const int node = blockIdx.x;
  const int f    = threadIdx.x;
  const float mean = zxhat[(size_t)node * (2 * FIN) + f];
  const float s    = zxhat[(size_t)node * (2 * FIN) + FIN + f];
  float sp;
  if (s > 20.0f)       sp = s;
  else if (s < -20.0f) sp = __expf(s);
  else                 sp = log1pf(__expf(s));
  const float sd = sp + 0.01f;
  const float ev = eps[(size_t)node * FIN + f];
  const float z  = mean + sd * ev;
  z_out[(size_t)node * FIN + f] = z;
  float term = 0.5f * (z * z - ev * ev) - __logf(sd);
#pragma unroll
  for (int off = 16; off > 0; off >>= 1)
    term += __shfl_xor(term, off, 32);
  if ((threadIdx.x & 31) == 0) partial[threadIdx.x >> 5] = term;
  __syncthreads();
  if (threadIdx.x == 0) {
    float blocksum = partial[0] + partial[1] + partial[2] + partial[3];
    atomicAdd(l2_acc, blocksum);
  }
}

// -------------------------------------------------------------------------
__global__ void gib_bias_relu_kernel(float* __restrict__ Y,
                                     const float* __restrict__ bias,
                                     long long n_elems, int F) {
  long long i = (long long)blockIdx.x * blockDim.x + threadIdx.x;
  if (i >= n_elems) return;
  float v = Y[i] + bias[i % F];
  Y[i] = v > 0.0f ? v : 0.0f;
}

// -------------------------------------------------------------------------
__global__ void gib_logsoftmax_kernel(const float* __restrict__ acc,
                                      const float* __restrict__ b1,
                                      float* __restrict__ out, int n_nodes) {
  int node = blockIdx.x * blockDim.x + threadIdx.x;
  if (node >= n_nodes) return;
  float t[FOUT];
  float m = -3.4e38f;
#pragma unroll
  for (int c = 0; c < FOUT; ++c) {
    t[c] = acc[(size_t)node * FOUT + c] + b1[c];
    m = t[c] > m ? t[c] : m;
  }
  float s = 0.0f;
#pragma unroll
  for (int c = 0; c < FOUT; ++c) s += __expf(t[c] - m);
  float lse = m + __logf(s);
#pragma unroll
  for (int c = 0; c < FOUT; ++c)
    out[(size_t)node * FOUT + c] = t[c] - lse;
}

__global__ void gib_finalize_l2_kernel(const float* __restrict__ l2_acc,
                                       float* __restrict__ out_scalar, int n) {
  if (threadIdx.x == 0 && blockIdx.x == 0)
    out_scalar[0] = l2_acc[0] / (float)n;
}

// -------------------------------------------------------------------------
extern "C" void kernel_launch(void* const* d_in, const int* in_sizes, int n_in,
                              void* d_out, int out_size, void* d_ws, size_t ws_size,
                              hipStream_t stream) {
  const float* x       = (const float*)d_in[0];
  const int*   adj_row = (const int*)  d_in[1];
  const int*   adj_col = (const int*)  d_in[2];
  const float* adj_val = (const float*)d_in[3];
  const float* eps     = (const float*)d_in[4];
  const float* Wz1     = (const float*)d_in[5];
  const float* bz1     = (const float*)d_in[6];
  const float* Wz2     = (const float*)d_in[7];
  const float* bz2     = (const float*)d_in[8];
  const float* W0      = (const float*)d_in[9];
  const float* b0      = (const float*)d_in[10];
  const float* W1      = (const float*)d_in[11];
  const float* b1      = (const float*)d_in[12];

  const int       N = in_sizes[0] / FIN;    // 100000
  const long long E = in_sizes[1];          // 1600000
  float* out = (float*)d_out;

  // workspace layout (floats): N*720 + 1, then f16 transposed weights
  float* r0 = (float*)d_ws;                 // N*128 : feats_pre, then z/h
  float* r1 = r0 + (size_t)N * FIN;         // N*256 : feats, then t0 = h@W0
  float* r2 = r1 + (size_t)N * FHID;        // N*256 : z_x_hat, then h2
  float* r3 = r2 + (size_t)N * FHID;        // N*40  : t1 = h2@W1
  float* r4 = r3 + (size_t)N * FOUT;        // N*40  : out accumulator
  float* r5 = r4 + (size_t)N * FOUT;        // 1     : l2 accumulator
  _Float16* wz1t = (_Float16*)(r5 + 4);             // 128*128
  _Float16* wz2t = wz1t + (size_t)FIN * FIN;        // 128*256
  _Float16* w0t  = wz2t + (size_t)FIN * FHID;       // 128*256
  _Float16* w1t  = w0t  + (size_t)FIN * FHID;       // 256*40

  const int rowTiles = (N + 15) / 16;       // 6250
  const dim3 blk(256);
  const dim3 grid128((rowTiles + 7) / 8, (FIN  + 63) / 64);  // CT=4
  const dim3 grid256((rowTiles + 7) / 8, (FHID + 63) / 64);  // CT=4
  const dim3 grid40 ((rowTiles + 7) / 8, 1);                 // CT=3 (48 >= 40)

  auto zero = [&](float* p, long long n) {
    gib_zero_kernel<<<(unsigned)((n + 255) / 256), 256, 0, stream>>>(p, n);
  };

  // 0) weight transpose+f16 prep, zero accumulators (ws poisoned by harness)
  gib_w2h_kernel<<<(FIN * FIN + 255) / 256, 256, 0, stream>>>(Wz1, wz1t, FIN, FIN);
  gib_w2h_kernel<<<(FIN * FHID + 255) / 256, 256, 0, stream>>>(Wz2, wz2t, FIN, FHID);
  gib_w2h_kernel<<<(FIN * FHID + 255) / 256, 256, 0, stream>>>(W0, w0t, FIN, FHID);
  gib_w2h_kernel<<<(FHID * FOUT + 255) / 256, 256, 0, stream>>>(W1, w1t, FHID, FOUT);
  zero(r2, (long long)N * FHID);
  zero(r5, 1);

  // 1) feats_pre = relu(x @ Wz1 + bz1)        [N x 128]
  gib_gemm_wmma_kernel<4><<<grid128, blk, 0, stream>>>(x, wz1t, bz1, r0, N, FIN, FIN, 1);
  // 2) feats = feats_pre @ Wz2 + bz2          [N x 256]
  gib_gemm_wmma_kernel<4><<<grid256, blk, 0, stream>>>(r0, wz2t, bz2, r1, N, FIN, FHID, 0);
  // 3) z_x_hat = spmm(adj, feats)             [N x 256]
  {
    long long total = E * (FHID / 4);
    gib_spmm_atomic_kernel<<<(unsigned)((total + 255) / 256), 256, 0, stream>>>(
        adj_row, adj_col, adj_val, r1, r2, E, FHID / 4);
  }
  // 4) z = mean + std*eps ; l2 partials       [N x 128] -> r0
  gib_vae_kernel<<<N, 128, 0, stream>>>(r2, eps, r0, r5, N);

  // 5) t0 = z @ W0                            [N x 256] -> r1
  gib_gemm_wmma_kernel<4><<<grid256, blk, 0, stream>>>(r0, w0t, nullptr, r1, N, FIN, FHID, 0);
  // 6) h2 = relu(spmm(adj, t0) + b0)          [N x 256] -> r2
  zero(r2, (long long)N * FHID);
  {
    long long total = E * (FHID / 4);
    gib_spmm_atomic_kernel<<<(unsigned)((total + 255) / 256), 256, 0, stream>>>(
        adj_row, adj_col, adj_val, r1, r2, E, FHID / 4);
  }
  gib_bias_relu_kernel<<<(unsigned)(((long long)N * FHID + 255) / 256), 256, 0, stream>>>(
      r2, b0, (long long)N * FHID, FHID);

  // 7) t1 = h2 @ W1                           [N x 40] -> r3
  gib_gemm_wmma_kernel<3><<<grid40, blk, 0, stream>>>(r2, w1t, nullptr, r3, N, FHID, FOUT, 0);
  // 8) out_acc = spmm(adj, t1)                [N x 40] -> r4
  zero(r4, (long long)N * FOUT);
  {
    long long total = E * (FOUT / 4);
    gib_spmm_atomic_kernel<<<(unsigned)((total + 255) / 256), 256, 0, stream>>>(
        adj_row, adj_col, adj_val, r3, r4, E, FOUT / 4);
  }
  // 9) log_softmax(out_acc + b1) -> d_out[0 : N*40]
  gib_logsoftmax_kernel<<<(N + 255) / 256, 256, 0, stream>>>(r4, b1, out, N);
  // 10) l2 scalar -> d_out[N*40]
  gib_finalize_l2_kernel<<<1, 1, 0, stream>>>(r5, out + (size_t)N * FOUT, N);
}
